// Net_33861522162572
// MI455X (gfx1250) — compile-verified, matrix operand order checked by reading, not run on hardware
//
#include <hip/hip_runtime.h>
#include <hip/hip_bf16.h>

typedef __attribute__((ext_vector_type(16))) _Float16 v16h;
typedef __attribute__((ext_vector_type(8)))  _Float16 v8h;
typedef __attribute__((ext_vector_type(8)))  float    v8f;

union Frag16 { v16h v; v8h h[2]; };

#define B_ROWS  4096
#define HID     1024
#define IN_DIM  16
#define OUTD    5
#define LEVELS  17
#define PREVW   96
#define TOTW    (OUTD * LEVELS)   // 85

// LDS staging geometry for block_gemm: 3 stages x (A tile + B tile),
// each tile 128 rows x 32 halfs, padded to 40 halfs/row (80B) so the
// 16B fragment reads are LDS bank-conflict free.
#define KSTEP     32
#define ROWPAD    40
#define TILEHALF  (128 * ROWPAD)          // 5120 halfs = 10240 B per tile
#define STAGEHALF (2 * TILEHALF)          // A + B
#define NSTAGE    3

// ---------------------------------------------------------------------------
// Convert Wblk (3,1024,1024) f32 row-major -> WT (3,1024,1024) f16, transposed
// per matrix: WT[j][n][k] = Wblk[j][k][n]. Makes B-fragment loads contiguous.
// ---------------------------------------------------------------------------
__global__ void convert_wblkT(const float* __restrict__ Wblk, _Float16* __restrict__ WT) {
    size_t idx = (size_t)blockIdx.x * blockDim.x + threadIdx.x; // 3*2^20
    size_t j = idx >> 20;
    size_t r = idx & 0xFFFFFu;
    size_t n = r >> 10;
    size_t k = r & 1023u;
    WT[idx] = (_Float16)Wblk[(j << 20) + k * 1024 + n];
}

// ---------------------------------------------------------------------------
// prev[:, 0:16] = x
// ---------------------------------------------------------------------------
__global__ void init_prev(const float* __restrict__ x, float* __restrict__ prev) {
    int idx = blockIdx.x * blockDim.x + threadIdx.x;   // B_ROWS*16
    int m = idx >> 4, k = idx & 15;
    prev[(size_t)m * PREVW + k] = x[idx];
}

// ---------------------------------------------------------------------------
// Input layer: H0[m,n] = sum_k prev[m,k] * Win[k,n] + bin[n]   (fp32, K<=96)
// Output written as f16 for the WMMA block layers. No activation.
// grid = (HID/256, B_ROWS), block = 256
// ---------------------------------------------------------------------------
__global__ __launch_bounds__(256) void input_layer(
    const float* __restrict__ prev, const float* __restrict__ Win,
    const float* __restrict__ bin, _Float16* __restrict__ H0, int Kin)
{
    int n = blockIdx.x * blockDim.x + threadIdx.x;
    int m = blockIdx.y;
    const float* pr = prev + (size_t)m * PREVW;
    float acc = bin[n];
    for (int k = 0; k < Kin; ++k)
        acc = fmaf(pr[k], Win[(size_t)k * HID + n], acc);
    H0[(size_t)m * HID + n] = (_Float16)acc;
}

// ---------------------------------------------------------------------------
// Block layer GEMM: Out = relu(A(4096x1024,f16) * W(1024x1024) + bias) as f16
// W supplied transposed (WT[n][k], f16). f32 accumulation via WMMA.
// Workgroup = 8 waves, tile 128x128. Wave tile 64x32 = 4x2 WMMA 16x16 accs.
//
// Data movement: triple-buffered ASYNC copy (global_load_async_to_lds_b128,
// ASYNCcnt) of the 128x32 A and B tiles into LDS; one barrier per K-step.
// Stage `it` overwrites the buffer consumed at `it-1` -- safe because every
// wave past the iteration-`it` barrier already fed those LDS reads to WMMA.
//
// Fragment layout (ISA 7.12.2, 16-bit A 16x32): lane holds row M=lane%16,
// elems 0..7 = K[koff..koff+7], elems 8..15 = K[16+koff..], koff=8*(lane/16).
// C/D layout: reg r -> M = r + 8*(lane/16), N = lane%16.
// ---------------------------------------------------------------------------
__global__ __launch_bounds__(256) void block_gemm(
    const _Float16* __restrict__ A, const _Float16* __restrict__ WT,
    const float* __restrict__ bias, _Float16* __restrict__ Out)
{
    const int K = HID, N = HID;
    __shared__ _Float16 smem[NSTAGE * STAGEHALF];   // 60 KB of the 320 KB WGP LDS

    const int tileM = blockIdx.x * 128;
    const int tileN = blockIdx.y * 128;
    const int tid  = threadIdx.x;
    const int wave = tid >> 5;
    const int lane = tid & 31;
    const int wr = wave >> 2;        // 0..1
    const int wc = wave & 3;         // 0..3
    const int lrow = lane & 15;
    const int koff = (lane >> 4) * 8;

    // Low 32 bits of the generic LDS address are the wave-relative LDS byte
    // offset (ISA 10.2 aperture rules) -- what the async VDST operand wants.
    const unsigned ldsBase = (unsigned)(size_t)(&smem[0]);

    // Issue one stage: 256 threads x 2 reps x (A + B) 16-byte async chunks.
    // 4 async instructions per wave per stage.
    auto stage = [&](int buf, int kk) {
#pragma unroll
        for (int rep = 0; rep < 2; ++rep) {
            const int ch  = tid + rep * 256;     // 0..511
            const int row = ch >> 2;             // 0..127
            const int sub = ch & 3;              // 16B chunk within 64B row
            const _Float16* gA = A  + (size_t)(tileM + row) * K + kk + sub * 8;
            const _Float16* gB = WT + (size_t)(tileN + row) * K + kk + sub * 8;
            const unsigned la = ldsBase +
                (unsigned)(buf * (STAGEHALF * 2) + (row * ROWPAD + sub * 8) * 2);
            const unsigned lb = la + TILEHALF * 2;
            asm volatile("global_load_async_to_lds_b128 %0, %1, off"
                         :: "v"(la), "v"(gA) : "memory");
            asm volatile("global_load_async_to_lds_b128 %0, %1, off"
                         :: "v"(lb), "v"(gB) : "memory");
        }
    };

    v8f acc[4][2] = {};

    stage(0, 0);
    stage(1, KSTEP);

    const int NIT = K / KSTEP;
    for (int it = 0; it < NIT; ++it) {
        const int kk  = it * KSTEP;
        const int cur = it % NSTAGE;

        // Oldest stage must be complete; newest (4 ops) may stay in flight.
        if (it < NIT - 1) asm volatile("s_wait_asynccnt 4" ::: "memory");
        else              asm volatile("s_wait_asynccnt 0" ::: "memory");
        __syncthreads();

        if (kk + 2 * KSTEP < K) stage((cur + 2) % NSTAGE, kk + 2 * KSTEP);

        const _Float16* As = smem + cur * STAGEHALF;
        const _Float16* Bs = As + TILEHALF;

        Frag16 a[4], b[2];
#pragma unroll
        for (int mt = 0; mt < 4; ++mt) {
            const _Float16* p = As + (wr * 64 + mt * 16 + lrow) * ROWPAD + koff;
            a[mt].h[0] = *(const v8h*)(p);
            a[mt].h[1] = *(const v8h*)(p + 16);
        }
#pragma unroll
        for (int nt = 0; nt < 2; ++nt) {
            const _Float16* p = Bs + (wc * 32 + nt * 16 + lrow) * ROWPAD + koff;
            b[nt].h[0] = *(const v8h*)(p);
            b[nt].h[1] = *(const v8h*)(p + 16);
        }
#pragma unroll
        for (int mt = 0; mt < 4; ++mt)
#pragma unroll
            for (int nt = 0; nt < 2; ++nt)
                acc[mt][nt] = __builtin_amdgcn_wmma_f32_16x16x32_f16(
                    false, a[mt].v, false, b[nt].v,
                    (short)0, acc[mt][nt], false, false);
    }

    const int mbase = tileM + wr * 64;
    const int nbase = tileN + wc * 32;
    const int rsel  = (lane >> 4) * 8;
#pragma unroll
    for (int mt = 0; mt < 4; ++mt) {
#pragma unroll
        for (int nt = 0; nt < 2; ++nt) {
            const int n = nbase + nt * 16 + lrow;
            const float bv = bias[n];
#pragma unroll
            for (int r = 0; r < 8; ++r) {
                const int m = mbase + mt * 16 + rsel + r;
                float v = acc[mt][nt][r] + bv;
                v = v > 0.0f ? v : 0.0f;          // relu (all 3 block layers)
                Out[(size_t)m * N + n] = (_Float16)v;
            }
        }
    }
}

// ---------------------------------------------------------------------------
// Output layer: out[m,d] = sum_k h[m,k]*Wout[k,d] + bout[d], d<5, K=1024.
// One wave per row m; writes into tot[m, level*5+d] and prev[m, 16+level*5+d].
// grid = B_ROWS/8, block = 256 (8 waves)
// ---------------------------------------------------------------------------
__global__ __launch_bounds__(256) void out_layer(
    const _Float16* __restrict__ Hh, const float* __restrict__ Wout,
    const float* __restrict__ bout, float* __restrict__ tot,
    float* __restrict__ prev, int level)
{
    const int wave = threadIdx.x >> 5, lane = threadIdx.x & 31;
    const int m = blockIdx.x * 8 + wave;
    const _Float16* hr = Hh + (size_t)m * HID;
    float acc[OUTD] = {0.f, 0.f, 0.f, 0.f, 0.f};
    for (int k = lane; k < HID; k += 32) {
        const float hv = (float)hr[k];
#pragma unroll
        for (int d = 0; d < OUTD; ++d)
            acc[d] = fmaf(hv, Wout[k * OUTD + d], acc[d]);
    }
#pragma unroll
    for (int d = 0; d < OUTD; ++d)
#pragma unroll
        for (int off = 16; off > 0; off >>= 1)
            acc[d] += __shfl_xor(acc[d], off, 32);
    if (lane < OUTD) {
        const float v = acc[lane] + bout[lane];
        tot [(size_t)m * TOTW  + level * OUTD + lane] = v;
        prev[(size_t)m * PREVW + IN_DIM + level * OUTD + lane] = v;
    }
}

// ---------------------------------------------------------------------------
// Attention + final projection. One block (96 threads) per batch row.
// tf[d,t]=tot_f[d*17+t]; tb[d,t]=tot_b[d*17+(16-t)]; softmax over t (axis=1).
// ---------------------------------------------------------------------------
__global__ __launch_bounds__(96) void attn_final(
    const float* __restrict__ tot_f, const float* __restrict__ tot_b,
    const float* __restrict__ W1, const float* __restrict__ W2,
    const float* __restrict__ W3, const float* __restrict__ W4,
    const float* __restrict__ b4, float* __restrict__ out)
{
    __shared__ float tf[TOTW], tb[TOTW];
    __shared__ float q[LEVELS][OUTD], kk[OUTD][LEVELS], vv[LEVELS][OUTD];
    __shared__ float sc[LEVELS][LEVELS], att[LEVELS][LEVELS], res[TOTW];
    __shared__ float w1[25], w2[25], w3[25];

    const int b = blockIdx.x, t = threadIdx.x;

    if (t < TOTW) {
        tf[t] = tot_f[(size_t)b * TOTW + t];
        const int d = t / LEVELS, s = t % LEVELS;
        tb[d * LEVELS + s] = tot_b[(size_t)b * TOTW + d * LEVELS + (LEVELS - 1 - s)];
    }
    if (t < 25) { w1[t] = W1[t]; w2[t] = W2[t]; w3[t] = W3[t]; }
    __syncthreads();

    if (t < TOTW) {
        const int tt = t / OUTD, e = t % OUTD;
        float qa = 0.f, ka = 0.f, va = 0.f;
#pragma unroll
        for (int d = 0; d < OUTD; ++d) {
            qa = fmaf(tf[d * LEVELS + tt], w1[d * OUTD + e], qa);
            ka = fmaf(tb[d * LEVELS + tt], w2[d * OUTD + e], ka);
            va = fmaf(tb[d * LEVELS + tt], w3[d * OUTD + e], va);
        }
        q[tt][e] = qa; kk[e][tt] = ka; vv[tt][e] = va;
    }
    __syncthreads();

    for (int i = t; i < LEVELS * LEVELS; i += 96) {
        const int tt = i / LEVELS, s = i % LEVELS;
        float a = 0.f;
#pragma unroll
        for (int e = 0; e < OUTD; ++e) a = fmaf(q[tt][e], kk[e][s], a);
        sc[tt][s] = a;
    }
    __syncthreads();

    if (t < LEVELS) {                 // column s = t, softmax over tt
        float mx = -1e30f;
        for (int tt = 0; tt < LEVELS; ++tt) mx = fmaxf(mx, sc[tt][t]);
        float sum = 0.f;
        for (int tt = 0; tt < LEVELS; ++tt) {
            const float e = expf(sc[tt][t] - mx);
            att[tt][t] = e; sum += e;
        }
        const float inv = 1.0f / sum;
        for (int tt = 0; tt < LEVELS; ++tt) att[tt][t] *= inv;
    }
    __syncthreads();

    if (t < TOTW) {
        const int tt = t / OUTD, e = t % OUTD;
        float a = 0.f;
        for (int s = 0; s < LEVELS; ++s) a = fmaf(att[tt][s], vv[s][e], a);
        res[t] = a;                    // flat index tt*5+e == t
    }
    __syncthreads();

    if (t < TOTW) {
        float a = b4[t];
        for (int j = 0; j < TOTW; ++j) a = fmaf(res[j], W4[j * TOTW + t], a);
        out[(size_t)b * TOTW + t] = a;
    }
}

// ---------------------------------------------------------------------------
extern "C" void kernel_launch(void* const* d_in, const int* in_sizes, int n_in,
                              void* d_out, int out_size, void* d_ws, size_t ws_size,
                              hipStream_t stream)
{
    const float* x      = (const float*)d_in[0];
    const float* Win_f  = (const float*)d_in[1];
    const float* bin_f  = (const float*)d_in[2];
    const float* Win_b  = (const float*)d_in[3];
    const float* bin_b  = (const float*)d_in[4];
    const float* Wblk   = (const float*)d_in[5];
    const float* bblk   = (const float*)d_in[6];
    const float* Wout_f = (const float*)d_in[7];
    const float* bout_f = (const float*)d_in[8];
    const float* Wout_b = (const float*)d_in[9];
    const float* bout_b = (const float*)d_in[10];
    const float* W1     = (const float*)d_in[11];
    const float* W2     = (const float*)d_in[12];
    const float* W3     = (const float*)d_in[13];
    const float* W4     = (const float*)d_in[14];
    const float* b4     = (const float*)d_in[15];

    char* ws = (char*)d_ws;
    size_t off = 0;
    auto carve = [&](size_t bytes) -> char* {
        char* p = ws + off;
        off = (off + bytes + 255) & ~(size_t)255;
        return p;
    };
    _Float16* WT      = (_Float16*)carve((size_t)3 * HID * HID * sizeof(_Float16));
    _Float16* hA      = (_Float16*)carve((size_t)B_ROWS * HID * sizeof(_Float16));
    _Float16* hB      = (_Float16*)carve((size_t)B_ROWS * HID * sizeof(_Float16));
    float*    prevbuf = (float*)   carve((size_t)B_ROWS * PREVW * sizeof(float));
    float*    totf    = (float*)   carve((size_t)B_ROWS * TOTW * sizeof(float));
    float*    totb    = (float*)   carve((size_t)B_ROWS * TOTW * sizeof(float));
    (void)ws_size; (void)n_in; (void)in_sizes; (void)out_size;

    // one-time (per call) weight convert+transpose for the hot 1024x1024 GEMMs
    convert_wblkT<<<(3u << 20) / 256, 256, 0, stream>>>(Wblk, WT);

    const dim3 gGemm(B_ROWS / 128, HID / 128);   // 32 x 8

    for (int dir = 0; dir < 2; ++dir) {
        const float* Win  = dir ? Win_b  : Win_f;
        const float* bin  = dir ? bin_b  : bin_f;
        const float* Wout = dir ? Wout_b : Wout_f;
        const float* bout = dir ? bout_b : bout_f;
        float* tot = dir ? totb : totf;

        init_prev<<<(B_ROWS * IN_DIM) / 256, 256, 0, stream>>>(x, prevbuf);

        for (int i = 0; i < LEVELS; ++i) {
            const int Kin = IN_DIM + OUTD * i;
            input_layer<<<dim3(HID / 256, B_ROWS), 256, 0, stream>>>(
                prevbuf, Win + (size_t)i * PREVW * HID, bin + (size_t)i * HID, hA, Kin);

            block_gemm<<<gGemm, 256, 0, stream>>>(hA, WT,                     bblk,            hB);
            block_gemm<<<gGemm, 256, 0, stream>>>(hB, WT + ((size_t)1 << 20), bblk + HID,      hA);
            block_gemm<<<gGemm, 256, 0, stream>>>(hA, WT + ((size_t)2 << 20), bblk + 2 * HID,  hB);

            out_layer<<<B_ROWS / 8, 256, 0, stream>>>(
                hB, Wout + (size_t)i * HID * OUTD, bout + (size_t)i * OUTD,
                tot, prevbuf, i);
        }
    }

    attn_final<<<B_ROWS, 96, 0, stream>>>(totf, totb, W1, W2, W3, W4, b4, (float*)d_out);
}
